// KnowledgeSheaf_22471268893190
// MI455X (gfx1250) — matrix operand order repaired and checked by reading, not compiled, not run on hardware
//
#include <hip/hip_runtime.h>
#include <hip/hip_bf16.h>

typedef __attribute__((ext_vector_type(16))) _Float16 v16h;
typedef __attribute__((ext_vector_type(8)))  _Float16 v8h;
typedef __attribute__((ext_vector_type(8)))  float    v8f;

#define SD 16      // stalk dim
#define NT 8       // n types

// ---------------- zero scratch + output ----------------
__global__ void k_zero(unsigned* ws, long zwords, float* out) {
  long i = (long)blockIdx.x * blockDim.x + threadIdx.x;
  if (i < zwords) ws[i] = 0u;
  if (i == 0) out[0] = 0.f;
}

// ---------------- in-degree over dst ----------------
__global__ void k_deg(const int* __restrict__ dstIdx, unsigned* __restrict__ deg, int nE) {
  int e = blockIdx.x * blockDim.x + threadIdx.x;
  if (e < nE) atomicAdd(&deg[dstIdx[e]], 1u);
}

// ---------------- transpose + scale + f16 convert: xT[n][k] ----------------
__global__ void k_prep_x(const float* __restrict__ reps, const unsigned* __restrict__ deg,
                         _Float16* __restrict__ xT, int nN) {
  int n = blockIdx.x * blockDim.x + threadIdx.x;
  if (n < nN) {
    unsigned d = deg[n];
    float s = d ? rsqrtf((float)d) : 0.f;
#pragma unroll
    for (int k = 0; k < SD; ++k)
      xT[(size_t)n * SD + k] = (_Float16)(reps[(size_t)k * nN + n] * s);
  }
}

// ---------------- restriction maps -> f16 table ----------------
__global__ void k_prep_maps(const float* __restrict__ rmaps, _Float16* __restrict__ mapsH,
                            int n) {
  int i = blockIdx.x * blockDim.x + threadIdx.x;
  if (i < n) mapsH[i] = (_Float16)rmaps[i];
}

// ---------------- bucket edges by (type_src, type_dst) ----------------
__global__ void k_count(const int* __restrict__ src, const int* __restrict__ dst,
                        const int* __restrict__ types, unsigned* __restrict__ key,
                        unsigned* __restrict__ cnt, int nE) {
  int e = blockIdx.x * blockDim.x + threadIdx.x;
  if (e < nE) {
    unsigned k = (unsigned)(types[src[e]] * NT + types[dst[e]]);
    key[e] = k;
    atomicAdd(&cnt[k], 1u);
  }
}

__global__ void k_scan(const unsigned* __restrict__ cnt, unsigned* __restrict__ cur) {
  if (threadIdx.x == 0 && blockIdx.x == 0) {
    unsigned acc = 0;
    for (int i = 0; i < NT * NT; ++i) { cur[i] = acc; acc += cnt[i]; }
  }
}

__global__ void k_bin(const unsigned* __restrict__ key, unsigned* __restrict__ cur,
                      unsigned* __restrict__ binned, int nE) {
  int e = blockIdx.x * blockDim.x + threadIdx.x;
  if (e < nE) {
    unsigned p = atomicAdd(&cur[key[e]], 1u);
    binned[p] = (unsigned)e;
  }
}

// ---------------- WMMA edge kernel ----------------
// One wave per tile of 16 binned edges.
// B (32x16 f16): col n = scaled f16 x-vector of edge n's endpoint (K=0..15),
//                K=16..31 zero (lanes 16..31 hold zeros; this code uses k = 16h + s).
// A (16x32 f16): shared per-bucket restriction map, K zero-padded.
// D (16x16 f32): lane L<16 -> rows 0..7 of col L; lane L+16 -> rows 8..15.
__global__ __launch_bounds__(32) void k_gemm(
    const _Float16* __restrict__ xT, const _Float16* __restrict__ mapsH,
    const int* __restrict__ src, const int* __restrict__ dst,
    const int* __restrict__ types, const unsigned* __restrict__ binned,
    float* __restrict__ comp, int nE) {
  const int lane = threadIdx.x;
  const int col  = lane & 15;
  const int half = lane >> 4;          // 0: rows 0..7, K 0..15 ; 1: rows 8..15, K 16..31 (zero)
  const int pos  = blockIdx.x * 16 + col;
  const bool haveEdge = pos < nE;

  __builtin_prefetch(binned + pos + 16, 0, 0);   // global_prefetch_b8

  unsigned e = 0; int sN = 0, dN = 0; unsigned key = 0xFFFFu;
  if (haveEdge) {
    e  = binned[pos];
    sN = src[e];
    dN = dst[e];
    key = (unsigned)(types[sN] * NT + types[dN]);
  }

  const v16h z16 = {};
  v16h bh = {}, bt = {};
  if (haveEdge && half == 0) {
    bh = *(const v16h*)(xT + (size_t)sN * SD);   // 32B contiguous gather
    bt = *(const v16h*)(xT + (size_t)dN * SD);
  }

  // wave-uniform loop over distinct type-pair keys present in this tile
  unsigned remaining = (unsigned)__ballot(haveEdge && half == 0) & 0xFFFFu;
  while (remaining) {
    int first = __ffs((int)remaining) - 1;
    unsigned curKey = (unsigned)__shfl((int)key, first);
    unsigned match = (unsigned)__ballot(half == 0 && haveEdge && key == curKey) & 0xFFFFu;
    remaining &= ~match;
    const bool inSeg = ((match >> col) & 1u) != 0u;

    v16h bhm = inSeg ? bh : z16;
    v16h btm = inSeg ? bt : z16;

    const int a = (int)(curKey >> 3), b = (int)(curKey & 7);
    const int row = col, k0 = half * 8;
    v16h Ah = {}, At = {};
    if (k0 == 0) {
      Ah = *(const v16h*)(mapsH + (size_t)(a * NT + b) * (SD * SD) + row * SD);
      At = *(const v16h*)(mapsH + (size_t)(b * NT + a) * (SD * SD) + row * SD);
    }

    v8f cz = {};
    // EXEC is all-ones here (loop condition is wave-uniform) as WMMA requires.
    v8f Dh = __builtin_amdgcn_wmma_f32_16x16x32_f16(false, Ah, false, bhm,
                                                    (short)0, cz, false, false);
    v8f Dt = __builtin_amdgcn_wmma_f32_16x16x32_f16(false, At, false, btm,
                                                    (short)0, cz, false, false);

    if (inSeg) {
      float* cs = comp + (size_t)sN * SD + half * 8;   // 32B contiguous scatter
      float* cd = comp + (size_t)dN * SD + half * 8;
#pragma unroll
      for (int i = 0; i < 8; ++i) {
        atomicAdd(cs + i,  Dh[i]);    // +head_emb
        atomicAdd(cd + i, -Dt[i]);    // -tail_emb
      }
    }
  }
}

// ---------------- sum of squares reduction ----------------
__global__ void k_redux(const float* __restrict__ comp, float* __restrict__ out, long n) {
  __shared__ float s[256];
  float v = 0.f;
  for (long i = (long)blockIdx.x * blockDim.x + threadIdx.x; i < n;
       i += (long)gridDim.x * blockDim.x) {
    float c = comp[i];
    v += c * c;
  }
  s[threadIdx.x] = v;
  __syncthreads();
  for (int o = 128; o > 0; o >>= 1) {
    if ((int)threadIdx.x < o) s[threadIdx.x] += s[threadIdx.x + o];
    __syncthreads();
  }
  if (threadIdx.x == 0) atomicAdd(out, s[0]);
}

extern "C" void kernel_launch(void* const* d_in, const int* in_sizes, int n_in,
                              void* d_out, int out_size, void* d_ws, size_t ws_size,
                              hipStream_t stream) {
  const float* reps  = (const float*)d_in[0];   // [16, N]
  const float* rmaps = (const float*)d_in[1];   // [8, 8, 16, 16]
  const int*   eidx  = (const int*)d_in[2];     // [2, E]
  const int*   types = (const int*)d_in[3];     // [N]
  const int nN = in_sizes[3];
  const int nE = in_sizes[2] / 2;
  const int* src = eidx;
  const int* dst = eidx + nE;

  // workspace layout (4-byte words), 32B-align vector-loaded regions
  auto al = [](size_t w) { return (w + 7) & ~(size_t)7; };
  size_t W_DEG  = 0;                              // u32 [N]
  size_t W_COMP = al(W_DEG + (size_t)nN);         // f32 [N][16]
  size_t W_CNT  = W_COMP + (size_t)SD * nN;       // u32 [64]
  size_t W_CUR  = W_CNT + 64;                     // u32 [64]
  size_t ZWORDS = W_CUR + 64;                     // zeroed every launch
  size_t W_KEY  = al(ZWORDS);                     // u32 [E]
  size_t W_BIN  = W_KEY + (size_t)nE;             // u32 [E]
  size_t W_XT   = al(W_BIN + (size_t)nE);         // f16 [N][16] (8N words)
  size_t W_MAPS = W_XT + (size_t)8 * nN;          // f16 [64][256] (8192 words)

  unsigned* ws     = (unsigned*)d_ws;
  unsigned* deg    = ws + W_DEG;
  float*    comp   = (float*)(ws + W_COMP);
  unsigned* cnt    = ws + W_CNT;
  unsigned* cur    = ws + W_CUR;
  unsigned* key    = ws + W_KEY;
  unsigned* binned = ws + W_BIN;
  _Float16* xT     = (_Float16*)(ws + W_XT);
  _Float16* mapsH  = (_Float16*)(ws + W_MAPS);
  float*    out    = (float*)d_out;

  const int tb = 256;
  const int nMapElts = NT * NT * SD * SD;
  k_zero<<<(int)((ZWORDS + tb - 1) / tb), tb, 0, stream>>>(ws, (long)ZWORDS, out);
  k_deg<<<(nE + tb - 1) / tb, tb, 0, stream>>>(dst, deg, nE);
  k_prep_x<<<(nN + tb - 1) / tb, tb, 0, stream>>>(reps, deg, xT, nN);
  k_prep_maps<<<(nMapElts + tb - 1) / tb, tb, 0, stream>>>(rmaps, mapsH, nMapElts);
  k_count<<<(nE + tb - 1) / tb, tb, 0, stream>>>(src, dst, types, key, cnt, nE);
  k_scan<<<1, 1, 0, stream>>>(cnt, cur);
  k_bin<<<(nE + tb - 1) / tb, tb, 0, stream>>>(key, cur, binned, nE);
  k_gemm<<<(nE + 15) / 16, 32, 0, stream>>>(xT, mapsH, src, dst, types, binned, comp, nE);
  k_redux<<<1024, tb, 0, stream>>>(comp, out, (long)SD * nN);
}
